// MetaLearnerNetwork_31516470018724
// MI455X (gfx1250) — matrix operand order repaired and checked by reading, not compiled, not run on hardware
//
#include <hip/hip_runtime.h>
#include <hip/hip_bf16.h>
#include <math.h>

// Problem constants
#define Bsz   128
#define Tt    1024
#define Hh    256
#define G3H   768      // 3*H
#define YDm   30
#define NBLK  64
#define NTHR  256
#define NWAVE (NBLK * (NTHR / 32))
#define NTHREADS (NBLK * NTHR)

typedef __attribute__((ext_vector_type(16))) __bf16 v16bf;
typedef __attribute__((ext_vector_type(8)))  __bf16 v8bf;
typedef __attribute__((ext_vector_type(8)))  float  v8f;

// ---------------- workspace layout (bytes) ----------------
static constexpr size_t O_BAR  = 0;                       // 256
static constexpr size_t O_HBF  = 256;                     // 3*B*H bf16 (masked h)
static constexpr size_t O_XBF  = O_HBF  + 3*Bsz*Hh*2;     // 3*B*H bf16 (unmasked layer out)
static constexpr size_t O_HF32 = O_XBF  + 3*Bsz*Hh*2;     // 3*B*H f32 (masked h)
static constexpr size_t O_GI   = O_HF32 + 3*Bsz*Hh*4;     // B*768 f32
static constexpr size_t O_GH   = O_GI   + (size_t)Bsz*G3H*4;   // 3*B*768 f32
static constexpr size_t O_H2   = O_GH   + (size_t)3*Bsz*G3H*4; // B*H f32 (unmasked h2)
static constexpr size_t ZERO_BYTES = O_H2 + (size_t)Bsz*Hh*4;
static constexpr size_t O_X0   = ZERO_BYTES;              // T*B*32 bf16, flipped, padded
static constexpr size_t O_DF   = O_X0   + (size_t)Tt*Bsz*32*2; // T*B f32 done (flipped)
static constexpr size_t O_WIH0 = O_DF   + (size_t)Tt*Bsz*4;    // 768*32 bf16 (padded)
static constexpr size_t O_WIH1 = O_WIH0 + 768*32*2;
static constexpr size_t O_WIH2 = O_WIH1 + (size_t)G3H*Hh*2;
static constexpr size_t O_WHH0 = O_WIH2 + (size_t)G3H*Hh*2;
static constexpr size_t O_WHH1 = O_WHH0 + (size_t)G3H*Hh*2;
static constexpr size_t O_WHH2 = O_WHH1 + (size_t)G3H*Hh*2;

struct GruParams {
  const __bf16* x0;      // [T][B][32]
  const float*  df;      // [T][B]
  const __bf16* Wih0;    // [768][32]
  const __bf16* Wih1;    // [768][256]
  const __bf16* Wih2;
  const __bf16* Whh0;    // [768][256]
  const __bf16* Whh1;
  const __bf16* Whh2;
  const float *bih0, *bhh0, *bih1, *bhh1, *bih2, *bhh2;
  __bf16* hbf;           // [3][B][H] masked bf16
  __bf16* xbf;           // [3][B][H] unmasked bf16 (layer output -> next layer input)
  float*  hf32;          // [3][B][H] masked f32
  float*  GI;            // [B][768]
  float*  GH;            // [3][B][768]
  float*  h2;            // [B][H] unmasked last-layer output
  const float *wfy, *bfy, *wfp, *bfp;
  float*  out;           // d_out: pi[B*T] then yo[B*T*30]
  unsigned* bar;
};

__device__ __forceinline__ float sigf(float x) { return 1.0f / (1.0f + __expf(-x)); }

// ---------------- device-wide epoch barrier ----------------
__device__ __forceinline__ void gbar(unsigned* cnt, unsigned target) {
  __syncthreads();
  __threadfence();
  if (threadIdx.x == 0) {
    __hip_atomic_fetch_add(cnt, 1u, __ATOMIC_ACQ_REL, __HIP_MEMORY_SCOPE_AGENT);
    while (__hip_atomic_load(cnt, __ATOMIC_ACQUIRE, __HIP_MEMORY_SCOPE_AGENT) < target)
      __builtin_amdgcn_s_sleep(8);
  }
  __syncthreads();
  __threadfence();
}

// ---------------- one 16x16 WMMA output tile ----------------
// A: [rows][lda] bf16 row-major (activations).  W: [out_ch][ldw] bf16 row-major,
// so B-matrix column n == W row (nb+n), contiguous in K.  out: f32 [.. ][ldo].
__device__ __forceinline__ void wmma_tile(const __bf16* __restrict__ A, int lda,
                                          const __bf16* __restrict__ W, int ldw, int K,
                                          const float* __restrict__ bias,
                                          float* __restrict__ out, int ldo,
                                          int mb, int nb, int lane) {
  const int lm = lane & 15;
  const int hi = lane >> 4;
  v8f c = {0.f, 0.f, 0.f, 0.f, 0.f, 0.f, 0.f, 0.f};
  const __bf16* ap = A + (size_t)(mb + lm) * lda + hi * 16;  // per-lane A row base
  const __bf16* wp = W + (size_t)(nb + lm) * ldw + hi * 16;  // per-lane B column base
  for (int k = 0; k < K; k += 32) {
    v8bf alo = *(const v8bf*)(ap + k);        // K k+ (0..7 | 16..23)
    v8bf ahi = *(const v8bf*)(ap + k + 8);    // K k+ (8..15 | 24..31)
    v16bf a;
#pragma unroll
    for (int i = 0; i < 8; ++i) { a[i] = alo[i]; a[i + 8] = ahi[i]; }
    v16bf bv = *(const v16bf*)(wp + k);       // 16 consecutive K of one column
    c = __builtin_amdgcn_wmma_f32_16x16x32_bf16(false, a, false, bv,
                                                (short)0, c, false, false);
  }
  const float bs = bias[nb + lm];
  float* op = out + (size_t)(mb + hi * 8) * ldo + nb + lm;
#pragma unroll
  for (int g = 0; g < 8; ++g) op[(size_t)g * ldo] = c[g] + bs;
}

// ---------------- gate nonlinearity phase ----------------
__device__ __forceinline__ void gates(const GruParams& p, int l, int s, int gt) {
  for (int i = gt; i < Bsz * Hh; i += NTHREADS) {
    const int b = i >> 8, ch = i & 255;
    const float* GHl = p.GH + (size_t)l * Bsz * G3H + (size_t)b * G3H;
    const float* GIb = p.GI + (size_t)b * G3H;
    const float r = sigf(GIb[ch] + GHl[ch]);
    const float z = sigf(GIb[ch + 256] + GHl[ch + 256]);
    const float n = tanhf(GIb[ch + 512] + r * GHl[ch + 512]);
    const float hold = p.hf32[(size_t)l * Bsz * Hh + i];
    const float hn = (1.f - z) * n + z * hold;
    p.xbf[(size_t)l * Bsz * Hh + i] = (__bf16)hn;          // unmasked -> next layer input
    if (l == 2) p.h2[i] = hn;                              // unmasked -> head
    float mask = 1.f;
    if (s + 1 < Tt) mask = 1.f - p.df[(size_t)(s + 1) * Bsz + b];
    const float hm = hn * mask;                            // masked -> next step state
    p.hf32[(size_t)l * Bsz * Hh + i] = hm;
    p.hbf[(size_t)l * Bsz * Hh + i]  = (__bf16)hm;
  }
}

// ---------------- fused output heads (yo sigmoid + pi) ----------------
__device__ __forceinline__ void head(const GruParams& p, int sHead, int gt) {
  for (int o = gt; o < Bsz * (YDm + 1); o += NTHREADS) {
    const int b = o / (YDm + 1), j = o % (YDm + 1);
    const float* hrow = p.h2 + (size_t)b * Hh;
    const float* wrow; float bias;
    if (j < YDm) { wrow = p.wfy + (size_t)j * Hh; bias = p.bfy[j]; }
    else         { wrow = p.wfp;                  bias = p.bfp[0]; }
    float acc = bias;
#pragma unroll 4
    for (int k = 0; k < Hh; ++k) acc += hrow[k] * wrow[k];
    const int fi = Tt - 1 - sHead;  // un-flip
    if (j < YDm) p.out[(size_t)Bsz * Tt + ((size_t)b * Tt + fi) * YDm + j] = sigf(acc);
    else         p.out[(size_t)b * Tt + fi] = acc;
  }
}

// ---------------- persistent GRU scan kernel ----------------
__global__ void __launch_bounds__(NTHR) gru_kernel(GruParams p) {
  const int lane = threadIdx.x & 31;
  const int wv = blockIdx.x * (NTHR / 32) + (threadIdx.x >> 5);
  const int gt = blockIdx.x * NTHR + threadIdx.x;
  unsigned target = 0;

  for (int s = 0; s < Tt; ++s) {
    // Phase A: GH for all 3 layers (prev-step h) + GI layer0 + head(s-1)
    for (int job = wv; job < 4 * 384; job += NWAVE) {
      const int which = job / 384, tile = job % 384;
      const int mb = (tile & 7) * 16, nb = (tile >> 3) * 16;
      if (which < 3) {
        const __bf16* W = (which == 0) ? p.Whh0 : (which == 1) ? p.Whh1 : p.Whh2;
        const float* bias = (which == 0) ? p.bhh0 : (which == 1) ? p.bhh1 : p.bhh2;
        wmma_tile(p.hbf + (size_t)which * Bsz * Hh, Hh, W, Hh, Hh, bias,
                  p.GH + (size_t)which * Bsz * G3H, G3H, mb, nb, lane);
      } else {
        wmma_tile(p.x0 + (size_t)s * Bsz * 32, 32, p.Wih0, 32, 32, p.bih0,
                  p.GI, G3H, mb, nb, lane);
      }
    }
    if (s > 0) head(p, s - 1, gt);
    gbar(p.bar, target += NBLK);

    gates(p, 0, s, gt);
    gbar(p.bar, target += NBLK);

    for (int job = wv; job < 384; job += NWAVE) {   // GI layer1
      const int mb = (job & 7) * 16, nb = (job >> 3) * 16;
      wmma_tile(p.xbf + 0 * Bsz * Hh, Hh, p.Wih1, Hh, Hh, p.bih1, p.GI, G3H, mb, nb, lane);
    }
    gbar(p.bar, target += NBLK);

    gates(p, 1, s, gt);
    gbar(p.bar, target += NBLK);

    for (int job = wv; job < 384; job += NWAVE) {   // GI layer2
      const int mb = (job & 7) * 16, nb = (job >> 3) * 16;
      wmma_tile(p.xbf + 1 * Bsz * Hh, Hh, p.Wih2, Hh, Hh, p.bih2, p.GI, G3H, mb, nb, lane);
    }
    gbar(p.bar, target += NBLK);

    gates(p, 2, s, gt);
    gbar(p.bar, target += NBLK);
  }
  head(p, Tt - 1, gt);
}

// ---------------- prep: embeds + flipped padded bf16 inputs ----------------
__global__ void prep_kernel(const float* rew, const float* done, const float* gamma,
                            const float* prob, const float* y, const float* y1,
                            const float* we1, const float* be1,
                            const float* we2, const float* be2,
                            __bf16* x0, float* df) {
  const int idx = blockIdx.x * blockDim.x + threadIdx.x;
  if (idx >= Bsz * Tt) return;
  const int b = idx / Tt, t = idx % Tt;
  const int s = Tt - 1 - t;
  float fy[2];
  const float* ys[2] = { y  + ((size_t)b * Tt + t) * YDm,
                         y1 + ((size_t)b * Tt + t) * YDm };
  for (int e = 0; e < 2; ++e) {
    float acc2 = be2[0];
    for (int j = 0; j < 16; ++j) {
      float a = be1[j];
      const float* w = we1 + j * YDm;
      for (int k = 0; k < YDm; ++k) a += ys[e][k] * w[k];
      acc2 += fmaxf(a, 0.f) * we2[j];
    }
    fy[e] = sigf(acc2);
  }
  __bf16* xp = x0 + ((size_t)s * Bsz + b) * 32;
  const float dn = done[(size_t)b * Tt + t];
  xp[0] = (__bf16)rew[(size_t)b * Tt + t];
  xp[1] = (__bf16)dn;
  xp[2] = (__bf16)gamma[0];
  xp[3] = (__bf16)prob[(size_t)b * Tt + t];
  xp[4] = (__bf16)fy[0];
  xp[5] = (__bf16)fy[1];
  for (int j = 6; j < 32; ++j) xp[j] = (__bf16)0.f;
  df[(size_t)s * Bsz + b] = dn;
}

// ---------------- weight convert to bf16 ----------------
__global__ void wcvt_kernel(const float* wih0, const float* whh0,
                            const float* wih1, const float* whh1,
                            const float* wih2, const float* whh2,
                            __bf16* Wih0, __bf16* Wih1, __bf16* Wih2,
                            __bf16* Whh0, __bf16* Whh1, __bf16* Whh2) {
  const int i = blockIdx.x * blockDim.x + threadIdx.x;
  const int stride = gridDim.x * blockDim.x;
  for (int j = i; j < G3H * 32; j += stride) {
    const int r = j >> 5, c = j & 31;
    Wih0[j] = (__bf16)(c < 6 ? wih0[r * 6 + c] : 0.f);
  }
  for (int j = i; j < G3H * Hh; j += stride) {
    Wih1[j] = (__bf16)wih1[j];  Wih2[j] = (__bf16)wih2[j];
    Whh0[j] = (__bf16)whh0[j];  Whh1[j] = (__bf16)whh1[j];  Whh2[j] = (__bf16)whh2[j];
  }
}

__global__ void zero_kernel(unsigned* p, size_t n) {
  size_t i = (size_t)blockIdx.x * blockDim.x + threadIdx.x;
  const size_t stride = (size_t)gridDim.x * blockDim.x;
  for (; i < n; i += stride) p[i] = 0u;
}

extern "C" void kernel_launch(void* const* d_in, const int* in_sizes, int n_in,
                              void* d_out, int out_size, void* d_ws, size_t ws_size,
                              hipStream_t stream) {
  (void)in_sizes; (void)n_in; (void)out_size; (void)ws_size;
  char* ws = (char*)d_ws;

  zero_kernel<<<256, 256, 0, stream>>>((unsigned*)ws, ZERO_BYTES / 4);

  wcvt_kernel<<<512, 256, 0, stream>>>(
      (const float*)d_in[6],  (const float*)d_in[7],
      (const float*)d_in[10], (const float*)d_in[11],
      (const float*)d_in[14], (const float*)d_in[15],
      (__bf16*)(ws + O_WIH0), (__bf16*)(ws + O_WIH1), (__bf16*)(ws + O_WIH2),
      (__bf16*)(ws + O_WHH0), (__bf16*)(ws + O_WHH1), (__bf16*)(ws + O_WHH2));

  prep_kernel<<<(Bsz * Tt + 127) / 128, 128, 0, stream>>>(
      (const float*)d_in[0], (const float*)d_in[1], (const float*)d_in[2],
      (const float*)d_in[3], (const float*)d_in[4], (const float*)d_in[5],
      (const float*)d_in[22], (const float*)d_in[23],
      (const float*)d_in[24], (const float*)d_in[25],
      (__bf16*)(ws + O_X0), (float*)(ws + O_DF));

  GruParams p;
  p.x0   = (const __bf16*)(ws + O_X0);
  p.df   = (const float*)(ws + O_DF);
  p.Wih0 = (const __bf16*)(ws + O_WIH0);
  p.Wih1 = (const __bf16*)(ws + O_WIH1);
  p.Wih2 = (const __bf16*)(ws + O_WIH2);
  p.Whh0 = (const __bf16*)(ws + O_WHH0);
  p.Whh1 = (const __bf16*)(ws + O_WHH1);
  p.Whh2 = (const __bf16*)(ws + O_WHH2);
  p.bih0 = (const float*)d_in[8];  p.bhh0 = (const float*)d_in[9];
  p.bih1 = (const float*)d_in[12]; p.bhh1 = (const float*)d_in[13];
  p.bih2 = (const float*)d_in[16]; p.bhh2 = (const float*)d_in[17];
  p.hbf  = (__bf16*)(ws + O_HBF);
  p.xbf  = (__bf16*)(ws + O_XBF);
  p.hf32 = (float*)(ws + O_HF32);
  p.GI   = (float*)(ws + O_GI);
  p.GH   = (float*)(ws + O_GH);
  p.h2   = (float*)(ws + O_H2);
  p.wfy  = (const float*)d_in[18]; p.bfy = (const float*)d_in[19];
  p.wfp  = (const float*)d_in[20]; p.bfp = (const float*)d_in[21];
  p.out  = (float*)d_out;
  p.bar  = (unsigned*)(ws + O_BAR);

  gru_kernel<<<NBLK, NTHR, 0, stream>>>(p);
}
